// Block_75179107549987
// MI455X (gfx1250) — matrix operand order repaired and checked by reading, not compiled, not run on hardware
//
#include <hip/hip_runtime.h>

// ---------------------------------------------------------------------------
// CDNA5 (gfx1250, wave32) transformer block, bf16 data path:
//   weights pre-converted fp32->bf16 once; LN / GEMM / attention produce bf16
//   activations so GEMM hot loops stream raw bf16 (half the HBM bytes, no
//   per-iteration v_cvt). Matrix math: v_wmma_f32_16x16x32_bf16 (f32 acc).
//   Global->LDS staging uses the Tensor Data Mover when available
//   (tensor_load_to_lds + s_wait_tensorcnt), else vectorized uint4 copies.
// Shapes hardcoded: B=4, T=2048, C=1024, H=16, D=64, FF=4096.
// Workspace: ~184 MB.
// ---------------------------------------------------------------------------

typedef __bf16 bf16;
typedef __bf16 v16bf __attribute__((ext_vector_type(16)));
typedef __bf16 v4bf  __attribute__((ext_vector_type(4)));
typedef float  v8f   __attribute__((ext_vector_type(8)));

__device__ __forceinline__ v8f wmma_bf16(v16bf a, v16bf b, v8f c) {
  return __builtin_amdgcn_wmma_f32_16x16x32_bf16(false, a, false, b, (short)0, c,
                                                 false, false);
}

// ---------------------------------------------------------------------------
// Tensor Data Mover (TDM) helpers — guarded so absence of the builtin still
// compiles (manual staging fallback). This toolchain exposes the 6-arg form:
//   (u32x4 g0, i32x8 g1, i32x4 g2, i32x4 g3, i32x8 g4, i32 cpol)
// ---------------------------------------------------------------------------
#if defined(__has_builtin)
#if __has_builtin(__builtin_amdgcn_tensor_load_to_lds)
#define HAVE_TDM 1
#endif
#endif

__device__ __forceinline__ void tensor_wait0() {
#if __has_builtin(__builtin_amdgcn_s_wait_tensorcnt)
  __builtin_amdgcn_s_wait_tensorcnt((short)0);
#else
  asm volatile("s_wait_tensorcnt 0x0" ::: "memory");
#endif
  asm volatile("" ::: "memory");
}

#ifdef HAVE_TDM
typedef unsigned int u32x4 __attribute__((ext_vector_type(4)));
typedef int          i32x8 __attribute__((ext_vector_type(8)));
typedef int          i32x4 __attribute__((ext_vector_type(4)));

// 2D bf16 tile load, global -> LDS, with optional LDS row padding.
//   tensor_w/h : tensor extent (elements / rows) for OOB clamping
//   tile_w/h   : tile extent (elements / rows)
//   stride     : tensor row stride in elements
//   pad_iv/amt : D# pad_interval / pad_amount encodings (0 amt = no padding)
__device__ __forceinline__ void tdm_load_2d(unsigned lds_off, const bf16* gptr,
                                            unsigned tensor_w, unsigned tensor_h,
                                            unsigned tile_w, unsigned tile_h,
                                            unsigned stride,
                                            unsigned pad_iv, unsigned pad_amt)
{
  const unsigned long long ga = (unsigned long long)(size_t)gptr;
  u32x4 g0;
  g0.x = 0x1u;                                            // count=1 (valid)
  g0.y = lds_off;                                         // lds_addr
  g0.z = (unsigned)(ga & 0xffffffffu);                    // global_addr lo
  g0.w = (unsigned)((ga >> 32) & 0x01ffffffu) | (2u << 30); // addr hi | type=2
  i32x8 g1;
  unsigned w0 = (1u << 16);                               // data_size = 2B
  if (pad_amt) w0 |= (1u << 20) | (pad_iv << 22) | (pad_amt << 25);
  g1[0] = (int)w0;
  g1[1] = (int)((tensor_w & 0xffffu) << 16);              // dim0[15:0]
  g1[2] = (int)(((tensor_w >> 16) & 0xffffu) | ((tensor_h & 0xffffu) << 16));
  g1[3] = (int)(((tensor_h >> 16) & 0xffffu) | (tile_w << 16));
  g1[4] = (int)(tile_h & 0xffffu);                        // tile_dim1 (tile_dim2=0)
  g1[5] = (int)stride;                                    // dim0_stride lo32
  g1[6] = 0;
  g1[7] = 0;
  const i32x4 z4 = {};
  const i32x8 z8 = {};
  __builtin_amdgcn_tensor_load_to_lds(g0, g1, z4, z4, z8, 0);
}
#endif

// ---------------------------------------------------------------------------
// fp32 -> bf16 bulk convert (weights pre-pass), grid-stride, float4 in.
// ---------------------------------------------------------------------------
__global__ __launch_bounds__(256)
void f32_to_bf16_kernel(const float* __restrict__ src, bf16* __restrict__ dst,
                        int n4)
{
  int i = blockIdx.x * blockDim.x + threadIdx.x;
  const int stride = gridDim.x * blockDim.x;
  for (; i < n4; i += stride) {
    const float4 f = *(const float4*)(src + (size_t)i * 4);
    v4bf o;
    o[0] = (bf16)f.x; o[1] = (bf16)f.y; o[2] = (bf16)f.z; o[3] = (bf16)f.w;
    *(v4bf*)(dst + (size_t)i * 4) = o;
  }
}

// ---------------------------------------------------------------------------
// LayerNorm: one 256-thread block per row (C=1024), bf16 output.
// ---------------------------------------------------------------------------
__global__ __launch_bounds__(256)
void ln_kernel(const float* __restrict__ x, const float* __restrict__ w,
               const float* __restrict__ bvec, bf16* __restrict__ out, int C)
{
  const int row = blockIdx.x;
  const int tid = threadIdx.x;
  const float* xr = x + (size_t)row * C;

  const float4 f = *(const float4*)(xr + tid * 4);
  float s = f.x + f.y + f.z + f.w;
  float q = f.x * f.x + f.y * f.y + f.z * f.z + f.w * f.w;
  for (int m = 1; m < 32; m <<= 1) {
    s += __shfl_xor(s, m, 32);
    q += __shfl_xor(q, m, 32);
  }
  __shared__ float sh_s[8], sh_q[8];
  const int wid = tid >> 5, lane = tid & 31;
  if (lane == 0) { sh_s[wid] = s; sh_q[wid] = q; }
  __syncthreads();
  float ts = 0.f, tq = 0.f;
#pragma unroll
  for (int i = 0; i < 8; ++i) { ts += sh_s[i]; tq += sh_q[i]; }

  const float mu  = ts / (float)C;
  const float var = tq / (float)C - mu * mu;
  const float inv = rsqrtf(var + 1e-5f);

  const float4 wv = *(const float4*)(w + tid * 4);
  const float4 bb = *(const float4*)(bvec + tid * 4);
  v4bf o;
  o[0] = (bf16)((f.x - mu) * inv * wv.x + bb.x);
  o[1] = (bf16)((f.y - mu) * inv * wv.y + bb.y);
  o[2] = (bf16)((f.z - mu) * inv * wv.z + bb.z);
  o[3] = (bf16)((f.w - mu) * inv * wv.w + bb.w);
  *(v4bf*)(out + (size_t)row * C + tid * 4) = o;
}

// ---------------------------------------------------------------------------
// Tiled bf16 WMMA GEMM:  Y[M,N] = epi(A[M,K] @ W[K,N] + bias [+ R])
// Block tile 128x128x32, 8 waves, wave tile 32x64 = 2x4 WMMA 16x16 tiles.
// A, W are bf16; bias/residual fp32; output bf16 or fp32 (OUTBF).
// ---------------------------------------------------------------------------
#define BM 128
#define BN 128
#define BK 32

template <bool RELU, bool RESID, bool OUTBF>
__global__ __launch_bounds__(256)
void gemm_bf16_wmma(const bf16* __restrict__ A, const bf16* __restrict__ W,
                    const float* __restrict__ bias, const float* __restrict__ R,
                    void* __restrict__ Yv, int M, int N, int K)
{
  __shared__ bf16 As[BM][BK + 8];   // stride 40 halves = 80 B (16B multiple)
  __shared__ bf16 Bs[BK][BN + 8];   // stride 136 halves = 272 B

  const int tid  = threadIdx.x;
  const int wid  = tid >> 5;
  const int lane = tid & 31;
  const int half = lane >> 4;
  const int l16  = lane & 15;

  const int m0 = blockIdx.y * BM;
  const int n0 = blockIdx.x * BN;
  const int wm = (wid >> 1) * 32;
  const int wn = (wid & 1) * 64;

#ifdef HAVE_TDM
  const unsigned ldsA = (unsigned)(size_t)&As[0][0];
  const unsigned ldsB = (unsigned)(size_t)&Bs[0][0];
#endif

  v8f acc[2][4] = {};

  for (int k0 = 0; k0 < K; k0 += BK) {
#ifdef HAVE_TDM
    if (tid < 32) {   // wave 0 issues DMA for the whole block tile
      // A tile: 128 rows x 32 elems; LDS rows padded 16 dwords data + 4 pad
      tdm_load_2d(ldsA, A + (size_t)m0 * K + k0,
                  (unsigned)K, (unsigned)M, BK, BM, (unsigned)K, 3u, 3u);
      // B tile: 32 rows x 128 elems; 64 dwords data + 4 pad
      tdm_load_2d(ldsB, W + (size_t)k0 * N + n0,
                  (unsigned)N, (unsigned)K, BN, BK, (unsigned)N, 5u, 3u);
      tensor_wait0();
    }
    __syncthreads();
#else
    // manual staging: 512 uint4 per tile, 2 per thread each
#pragma unroll
    for (int i = 0; i < 2; ++i) {
      const int idx = tid + i * 256;
      const int r = idx >> 2, c8 = (idx & 3) << 3;
      *(uint4*)&As[r][c8] = *(const uint4*)(A + (size_t)(m0 + r) * K + k0 + c8);
    }
#pragma unroll
    for (int i = 0; i < 2; ++i) {
      const int idx = tid + i * 256;
      const int r = idx >> 4, c8 = (idx & 15) << 3;
      *(uint4*)&Bs[r][c8] = *(const uint4*)(W + (size_t)(k0 + r) * N + n0 + c8);
    }
    if (k0 + BK < K) {
      __builtin_prefetch(A + (size_t)(m0 + (tid >> 1)) * K + (k0 + BK), 0, 1);
      __builtin_prefetch(W + (size_t)(k0 + BK + (tid >> 3)) * N + n0 + ((tid & 7) << 4), 0, 1);
    }
    __syncthreads();
#endif

    // fragments: A and B both use contiguous k = 16*half + s
    v16bf af[2], bfg[4];
#pragma unroll
    for (int mi = 0; mi < 2; ++mi) {
      const int row = wm + mi * 16 + l16;
#pragma unroll
      for (int j = 0; j < 8; ++j) {
        const int kk = (half << 4) + (j << 1);
        af[mi][2 * j]     = As[row][kk];
        af[mi][2 * j + 1] = As[row][kk + 1];
      }
    }
#pragma unroll
    for (int ni = 0; ni < 4; ++ni) {
      const int col = wn + ni * 16 + l16;
#pragma unroll
      for (int i = 0; i < 16; ++i)
        bfg[ni][i] = Bs[(half << 4) + i][col];
    }
#pragma unroll
    for (int mi = 0; mi < 2; ++mi)
#pragma unroll
      for (int ni = 0; ni < 4; ++ni)
        acc[mi][ni] = wmma_bf16(af[mi], bfg[ni], acc[mi][ni]);
    __syncthreads();
  }

  // epilogue: bias [+relu] [+residual]; C/D layout M = r + 8*half
#pragma unroll
  for (int mi = 0; mi < 2; ++mi) {
#pragma unroll
    for (int ni = 0; ni < 4; ++ni) {
      const int gn = n0 + wn + ni * 16 + l16;
      const float bb = bias[gn];
#pragma unroll
      for (int r = 0; r < 8; ++r) {
        const int gm = m0 + wm + mi * 16 + r + (half << 3);
        float v = acc[mi][ni][r] + bb;
        if (RELU)  v = fmaxf(v, 0.0f);
        if (RESID) v += R[(size_t)gm * N + gn];
        if (OUTBF) ((bf16*)Yv)[(size_t)gm * N + gn] = (bf16)v;
        else       ((float*)Yv)[(size_t)gm * N + gn] = v;
      }
    }
  }
}

// ---------------------------------------------------------------------------
// Flash attention, causal, bf16 Q/K/V in, bf16 out.
// One wave per 16-row Q tile; KV chunks of 32; TDM stages per-wave tiles.
// ---------------------------------------------------------------------------
#define ATW 4

__global__ __launch_bounds__(32 * ATW)
void flash_attn_wmma(const bf16* __restrict__ Q, const bf16* __restrict__ Km,
                     const bf16* __restrict__ V, bf16* __restrict__ Y,
                     int Bn, int T, int Hh)
{
  __shared__ bf16 Qs[ATW][16][80];   // row stride 160 B
  __shared__ bf16 Ks[ATW][32][80];
  __shared__ bf16 Vs[ATW][32][80];
  __shared__ bf16 Ps[ATW][16][40];

  const int wid  = threadIdx.x >> 5;
  const int lane = threadIdx.x & 31;
  const int half = lane >> 4;
  const int l16  = lane & 15;

  const int tilesPerBH = T >> 4;
  const int qt = blockIdx.x * ATW + wid;
  const int bh = qt / tilesPerBH;
  const int q0 = (qt % tilesPerBH) << 4;
  const int b  = bh / Hh;
  const int h  = bh % Hh;
  const int C  = Hh * 64;

  const bf16* qp = Q  + ((size_t)b * T) * C + h * 64;
  const bf16* kp = Km + ((size_t)b * T) * C + h * 64;
  const bf16* vp = V  + ((size_t)b * T) * C + h * 64;

  // ---- stage Q tile (16x64) ----
#ifdef HAVE_TDM
  // rows: 32 dwords data + 8 dwords pad -> stride 80 halves
  tdm_load_2d((unsigned)(size_t)&Qs[wid][0][0], qp + (size_t)q0 * C,
              64u, (unsigned)T, 64u, 16u, (unsigned)C, 4u, 7u);
  tensor_wait0();
#else
#pragma unroll
  for (int c = 0; c < 4; ++c) {
    const int col = (half << 5) + (c << 3);
    *(uint4*)&Qs[wid][l16][col] = *(const uint4*)(qp + (size_t)(q0 + l16) * C + col);
  }
  asm volatile("s_wait_dscnt 0" ::: "memory");
#endif

  v16bf aq[2];
#pragma unroll
  for (int ks = 0; ks < 2; ++ks)
#pragma unroll
    for (int j = 0; j < 8; ++j) {
      const int kk = (ks << 5) + (half << 4) + (j << 1);
      aq[ks][2 * j]     = Qs[wid][l16][kk];
      aq[ks][2 * j + 1] = Qs[wid][l16][kk + 1];
    }

  v8f o[4] = {};
  float mrow[8], lrow[8];
#pragma unroll
  for (int r = 0; r < 8; ++r) { mrow[r] = -3.0e38f; lrow[r] = 0.0f; }

  const int kvEnd = q0 + 16;    // causal bound, wave-uniform
  for (int j0 = 0; j0 < kvEnd; j0 += 32) {
    // ---- stage K,V chunk (32x64) ----
#ifdef HAVE_TDM
    tdm_load_2d((unsigned)(size_t)&Ks[wid][0][0], kp + (size_t)j0 * C,
                64u, (unsigned)T, 64u, 32u, (unsigned)C, 4u, 7u);
    tdm_load_2d((unsigned)(size_t)&Vs[wid][0][0], vp + (size_t)j0 * C,
                64u, (unsigned)T, 64u, 32u, (unsigned)C, 4u, 7u);
    tensor_wait0();
#else
    const bf16* krow = kp + (size_t)(j0 + lane) * C;
    const bf16* vrow = vp + (size_t)(j0 + lane) * C;
#pragma unroll
    for (int c = 0; c < 8; ++c) {
      *(uint4*)&Ks[wid][lane][c << 3] = *(const uint4*)(krow + (c << 3));
      *(uint4*)&Vs[wid][lane][c << 3] = *(const uint4*)(vrow + (c << 3));
    }
    asm volatile("s_wait_dscnt 0" ::: "memory");
#endif

    // S(16x32) = Q K^T
    v8f s[2] = {};
#pragma unroll
    for (int nc = 0; nc < 2; ++nc)
#pragma unroll
      for (int ks = 0; ks < 2; ++ks) {
        v16bf bk;
#pragma unroll
        for (int i = 0; i < 16; ++i)
          bk[i] = Ks[wid][(nc << 4) + l16][(ks << 5) + (half << 4) + i];
        s[nc] = wmma_bf16(aq[ks], bk, s[nc]);
      }

    // scale + causal mask
#pragma unroll
    for (int nc = 0; nc < 2; ++nc)
#pragma unroll
      for (int r = 0; r < 8; ++r) {
        const float v = s[nc][r] * 0.125f;  // 1/sqrt(64)
        const int qrow = q0 + r + (half << 3);
        const int kcol = j0 + (nc << 4) + l16;
        s[nc][r] = (kcol > qrow) ? -1.0e30f : v;
      }

    // online softmax (rows live across a 16-lane group)
    float alpha[8];
#pragma unroll
    for (int r = 0; r < 8; ++r) {
      float mx = fmaxf(s[0][r], s[1][r]);
      for (int msk = 1; msk < 16; msk <<= 1) mx = fmaxf(mx, __shfl_xor(mx, msk, 32));
      const float mnew = fmaxf(mrow[r], mx);
      alpha[r] = __expf(mrow[r] - mnew);
      mrow[r]  = mnew;
      const float p0 = __expf(s[0][r] - mnew);
      const float p1 = __expf(s[1][r] - mnew);
      s[0][r] = p0; s[1][r] = p1;
      float rs = p0 + p1;
      for (int msk = 1; msk < 16; msk <<= 1) rs += __shfl_xor(rs, msk, 32);
      lrow[r] = lrow[r] * alpha[r] + rs;
    }
#pragma unroll
    for (int nt = 0; nt < 4; ++nt)
#pragma unroll
      for (int r = 0; r < 8; ++r) o[nt][r] *= alpha[r];

    // P: C-layout -> LDS -> A fragment
#pragma unroll
    for (int nc = 0; nc < 2; ++nc)
#pragma unroll
      for (int r = 0; r < 8; ++r)
        Ps[wid][r + (half << 3)][(nc << 4) + l16] = (bf16)s[nc][r];
    asm volatile("s_wait_dscnt 0" ::: "memory");

    v16bf ap;
#pragma unroll
    for (int j = 0; j < 8; ++j) {
      const int kk = (half << 4) + (j << 1);
      ap[2 * j]     = Ps[wid][l16][kk];
      ap[2 * j + 1] = Ps[wid][l16][kk + 1];
    }
#pragma unroll
    for (int nt = 0; nt < 4; ++nt) {
      v16bf bvv;
#pragma unroll
      for (int i = 0; i < 16; ++i)
        bvv[i] = Vs[wid][(half << 4) + i][(nt << 4) + l16];
      o[nt] = wmma_bf16(ap, bvv, o[nt]);
    }
  }

  // normalize + store bf16
  float inv[8];
#pragma unroll
  for (int r = 0; r < 8; ++r) inv[r] = 1.0f / lrow[r];
#pragma unroll
  for (int nt = 0; nt < 4; ++nt)
#pragma unroll
    for (int r = 0; r < 8; ++r) {
      const int gm = q0 + r + (half << 3);
      const int gn = h * 64 + (nt << 4) + l16;
      Y[((size_t)b * T + gm) * C + gn] = (bf16)(o[nt][r] * inv[r]);
    }
}

// ---------------------------------------------------------------------------
// Launch pipeline.
// ---------------------------------------------------------------------------
extern "C" void kernel_launch(void* const* d_in, const int* in_sizes, int n_in,
                              void* d_out, int out_size, void* d_ws, size_t ws_size,
                              hipStream_t stream) {
  (void)in_sizes; (void)n_in; (void)out_size; (void)ws_size;

  const float* x     = (const float*)d_in[0];
  const float* ln1_w = (const float*)d_in[1];
  const float* ln1_b = (const float*)d_in[2];
  const float* Wq    = (const float*)d_in[3];
  const float* bq    = (const float*)d_in[4];
  const float* Wk    = (const float*)d_in[5];
  const float* bk    = (const float*)d_in[6];
  const float* Wv    = (const float*)d_in[7];
  const float* bv    = (const float*)d_in[8];
  const float* Wo    = (const float*)d_in[9];
  const float* bo    = (const float*)d_in[10];
  const float* ln2_w = (const float*)d_in[11];
  const float* ln2_b = (const float*)d_in[12];
  const float* W1    = (const float*)d_in[13];
  const float* b1    = (const float*)d_in[14];
  const float* W2    = (const float*)d_in[15];
  const float* b2    = (const float*)d_in[16];
  float* out = (float*)d_out;

  const int Bn = 4, T = 2048, C = 1024, Hh = 16, FF = 4096;
  const int M = Bn * T;                       // 8192
  const size_t S  = (size_t)M * C;            // 8388608 activations
  const size_t CC = (size_t)C * C;            // 1M weight elems
  const size_t CF = (size_t)C * FF;           // 4M weight elems

  // workspace carve-up (bytes)
  char* p = (char*)d_ws;
  bf16* Wqb = (bf16*)p; p += CC * 2;
  bf16* Wkb = (bf16*)p; p += CC * 2;
  bf16* Wvb = (bf16*)p; p += CC * 2;
  bf16* Wob = (bf16*)p; p += CC * 2;
  bf16* W1b = (bf16*)p; p += CF * 2;
  bf16* W2b = (bf16*)p; p += CF * 2;
  bf16* h1b = (bf16*)p; p += S * 2;           // later reused as y
  bf16* qb  = (bf16*)p; p += S * 2;           // later reused as h2
  bf16* kb  = (bf16*)p; p += S * 2;
  bf16* vb  = (bf16*)p; p += S * 2;
  float* x1 = (float*)p; p += S * 4;
  bf16* f1b = (bf16*)p;                       // M x FF
  bf16* yb  = h1b;
  bf16* h2b = qb;

  auto conv = [&](const float* s, bf16* d, size_t n) {
    const int n4 = (int)(n >> 2);
    int blocks = (n4 + 255) / 256;
    if (blocks > 2048) blocks = 2048;
    f32_to_bf16_kernel<<<blocks, 256, 0, stream>>>(s, d, n4);
  };
  conv(Wq, Wqb, CC); conv(Wk, Wkb, CC); conv(Wv, Wvb, CC); conv(Wo, Wob, CC);
  conv(W1, W1b, CF); conv(W2, W2b, CF);

  const dim3 gC(C / BN, M / BM);    // (8, 64)
  const dim3 gF(FF / BN, M / BM);   // (32, 64)

  ln_kernel<<<M, 256, 0, stream>>>(x, ln1_w, ln1_b, h1b, C);

  gemm_bf16_wmma<false, false, true ><<<gC, 256, 0, stream>>>(h1b, Wqb, bq, nullptr, qb,  M, C, C);
  gemm_bf16_wmma<false, false, true ><<<gC, 256, 0, stream>>>(h1b, Wkb, bk, nullptr, kb,  M, C, C);
  gemm_bf16_wmma<false, false, true ><<<gC, 256, 0, stream>>>(h1b, Wvb, bv, nullptr, vb,  M, C, C);

  const int qTiles = Bn * Hh * (T / 16);      // 8192
  flash_attn_wmma<<<qTiles / ATW, 32 * ATW, 0, stream>>>(qb, kb, vb, yb, Bn, T, Hh);

  gemm_bf16_wmma<false, true, false><<<gC, 256, 0, stream>>>(yb, Wob, bo, x, x1, M, C, C);

  ln_kernel<<<M, 256, 0, stream>>>(x1, ln2_w, ln2_b, h2b, C);

  gemm_bf16_wmma<true, false, true ><<<gF, 256, 0, stream>>>(h2b, W1b, b1, nullptr, f1b, M, FF, C);
  gemm_bf16_wmma<false, true, false><<<gC, 256, 0, stream>>>(f1b, W2b, b2, x1, out, M, C, FF);
}
